// UNetBlock_61254823576090
// MI455X (gfx1250) — matrix operand order repaired and checked by reading, not compiled, not run on hardware
//
#include <hip/hip_runtime.h>
#include <cstdint>

// ---------------------------------------------------------------------------
// Problem constants (match reference)
// ---------------------------------------------------------------------------
#define Bc     16
#define Lc     1024
#define LPc    1028      // L + 2*pad (pad=2 for K=5 'SAME')
#define NINc   128
#define NSCc   192
#define NOUTc  256
#define KTAPS  5
#define Ec     16
#define Hc     4
#define DTEXTc 768
#define EPSc   1e-5f

typedef __attribute__((ext_vector_type(16))) _Float16 v16h;
typedef __attribute__((ext_vector_type(8)))  _Float16 v8h;
typedef __attribute__((ext_vector_type(8)))  float    v8f;
typedef __attribute__((ext_vector_type(4)))  float    v4f;

__device__ __forceinline__ float mishf(float x) {
    float sp = (x > 20.f) ? x : log1pf(__expf(x));
    return x * tanhf(sp);
}

// Block-wide sum reduce, blockDim.x == 256, red is shared float[256].
__device__ __forceinline__ float block_reduce_256(float v, float* red) {
    int tid = threadIdx.x;
    red[tid] = v;
    __syncthreads();
    #pragma unroll
    for (int s = 128; s > 0; s >>= 1) {
        if (tid < s) red[tid] += red[tid + s];
        __syncthreads();
    }
    float r = red[0];
    __syncthreads();
    return r;
}

// ---------------------------------------------------------------------------
// Generic zero fill (u32 words)
// ---------------------------------------------------------------------------
__global__ void k_fill_zero(uint32_t* p, size_t n) {
    size_t i = (size_t)blockIdx.x * blockDim.x + threadIdx.x;
    size_t stride = (size_t)gridDim.x * blockDim.x;
    for (; i < n; i += stride) p[i] = 0u;
}

// ---------------------------------------------------------------------------
// Time embedding: te[b][128] = (mish(emb @ fc1^T + b1)) @ fc2^T + b2
// grid B, block 128
// ---------------------------------------------------------------------------
__global__ __launch_bounds__(128)
void k_time_embed(const float* __restrict__ time_table, const int* __restrict__ t,
                  const float* __restrict__ w1, const float* __restrict__ b1,
                  const float* __restrict__ w2, const float* __restrict__ b2,
                  float* __restrict__ te) {
    __shared__ float se[64];
    __shared__ float h1[128];
    int b = blockIdx.x, tid = threadIdx.x;
    int tt = t[b];
    if (tid < 64) se[tid] = time_table[tt * 64 + tid];
    __syncthreads();
    float a = b1[tid];
    #pragma unroll 4
    for (int i = 0; i < 64; ++i) a += se[i] * w1[tid * 64 + i];
    h1[tid] = mishf(a);
    __syncthreads();
    float o = b2[tid];
    #pragma unroll 4
    for (int i = 0; i < 128; ++i) o += h1[i] * w2[tid * 128 + i];
    te[b * 128 + tid] = o;
}

// ---------------------------------------------------------------------------
// Transpose x (B,128,L) -> padded f16 XTT [B][LP][128] (with +te) and
// f16 XT [B][L][128] (raw, for residual 1x1 conv).
// grid (L/32, 128/32, B), block 256, LDS tile transpose.
// ---------------------------------------------------------------------------
__global__ __launch_bounds__(256)
void k_prep_xt(const float* __restrict__ x, const float* __restrict__ te,
               _Float16* __restrict__ XTT, _Float16* __restrict__ XT) {
    __shared__ float tile[32][33];
    int b = blockIdx.z, c0 = blockIdx.y * 32, l0 = blockIdx.x * 32;
    int lx = threadIdx.x & 31, rw = threadIdx.x >> 5;   // rw in 0..7
    #pragma unroll
    for (int cc = 0; cc < 4; ++cc) {
        int c = c0 + rw + cc * 8;
        tile[rw + cc * 8][lx] = x[((size_t)(b * NINc + c)) * Lc + l0 + lx];
    }
    __syncthreads();
    int c = c0 + lx;
    float tev = te[b * NINc + c];
    #pragma unroll
    for (int ll = 0; ll < 4; ++ll) {
        int l = l0 + rw + ll * 8;
        float v = tile[lx][rw + ll * 8];
        XT [((size_t)b * Lc  + l)     * NINc + c] = (_Float16)v;
        XTT[((size_t)b * LPc + l + 2) * NINc + c] = (_Float16)(v + tev);
    }
}

// ---------------------------------------------------------------------------
// Conv weight reorder (O,I,K) f32 -> [k][O][I] f16
// ---------------------------------------------------------------------------
__global__ void k_wconv_f16(const float* __restrict__ W, _Float16* __restrict__ Wo,
                            int O, int I, int KT) {
    int idx = blockIdx.x * blockDim.x + threadIdx.x;     // over O*I
    if (idx >= O * I) return;
    for (int k = 0; k < KT; ++k)
        Wo[(size_t)k * O * I + idx] = (_Float16)W[(size_t)idx * KT + k];
}

// ---------------------------------------------------------------------------
// Self-attention composed weights:
//  task 0..2 : Wqkv[task] (16x192) = in_w @ w ;  bqkv = in_w @ b + in_b
//  task 3    : Wo (192x32, cols 16..31 zero) = o_w @ out_w ; bo = o_w@out_b + o_b
// grid 4, block 256 (192 active)
// ---------------------------------------------------------------------------
__global__ __launch_bounds__(256)
void k_attn_wcomb(const float* qw, const float* qb, const float* kw, const float* kb,
                  const float* vw, const float* vb,
                  const float* iwq, const float* ibq, const float* iwk, const float* ibk,
                  const float* iwv, const float* ibv,
                  const float* ow, const float* ob, const float* outw, const float* outb,
                  _Float16* __restrict__ WQKV, float* __restrict__ BQKV,
                  _Float16* __restrict__ WO, float* __restrict__ BO) {
    int task = blockIdx.x, tid = threadIdx.x;
    if (task < 3) {
        const float* w  = (task == 0) ? qw  : (task == 1) ? kw  : vw;
        const float* bb = (task == 0) ? qb  : (task == 1) ? kb  : vb;
        const float* iw = (task == 0) ? iwq : (task == 1) ? iwk : iwv;
        const float* ib = (task == 0) ? ibq : (task == 1) ? ibk : ibv;
        if (tid < NSCc) {
            for (int e = 0; e < Ec; ++e) {
                float acc = 0.f;
                #pragma unroll
                for (int e2 = 0; e2 < Ec; ++e2) acc += iw[e * Ec + e2] * w[e2 * NSCc + tid];
                WQKV[(size_t)task * Ec * NSCc + e * NSCc + tid] = (_Float16)acc;
            }
        }
        if (tid < Ec) {
            float acc = ib[tid];
            #pragma unroll
            for (int e2 = 0; e2 < Ec; ++e2) acc += iw[tid * Ec + e2] * bb[e2];
            BQKV[task * Ec + tid] = acc;
        }
    } else {
        if (tid < NSCc) {
            for (int e = 0; e < Ec; ++e) {
                float acc = 0.f;
                #pragma unroll
                for (int e2 = 0; e2 < Ec; ++e2) acc += ow[tid * Ec + e2] * outw[e2 * Ec + e];
                WO[tid * 32 + e] = (_Float16)acc;
            }
            for (int e = Ec; e < 32; ++e) WO[tid * 32 + e] = (_Float16)0.f;
            float acc = ob[tid];
            #pragma unroll
            for (int e2 = 0; e2 < Ec; ++e2) acc += ow[tid * Ec + e2] * outb[e2];
            BO[tid] = acc;
        }
    }
}

// ---------------------------------------------------------------------------
// Cross-attention collapses exactly (all keys identical -> uniform softmax ->
// per-head output == vh). ca_vec[b][192] = ((LN(vc(text)) v-path) out-proj).
// grid B, block 256 (192 active)
// ---------------------------------------------------------------------------
__global__ __launch_bounds__(256)
void k_ca_vec(const float* __restrict__ text,
              const float* vc_w, const float* vc_b, const float* cn_g, const float* cn_b,
              const float* v_w, const float* v_b, const float* in_wv, const float* in_bv,
              const float* out_w, const float* out_b, const float* o_w, const float* o_b,
              float* __restrict__ CAVEC) {
    __shared__ float red[256];
    __shared__ float combn[NSCc];
    __shared__ float t16a[Ec], t16b[Ec], t16c[Ec];
    int b = blockIdx.x, tid = threadIdx.x;
    float comb = 0.f;
    if (tid < NSCc) {
        comb = vc_b[tid];
        for (int d = 0; d < DTEXTc; ++d) comb += text[b * DTEXTc + d] * vc_w[tid * DTEXTc + d];
    }
    float s = block_reduce_256((tid < NSCc) ? comb : 0.f, red);
    float mean = s / NSCc;
    float dv = (tid < NSCc) ? (comb - mean) : 0.f;
    float vs = block_reduce_256(dv * dv, red);
    float inv = rsqrtf(vs / NSCc + EPSc);
    if (tid < NSCc) combn[tid] = (comb - mean) * inv * cn_g[tid] + cn_b[tid];
    __syncthreads();
    if (tid < Ec) {
        float acc = v_b[tid];
        for (int c = 0; c < NSCc; ++c) acc += combn[c] * v_w[tid * NSCc + c];
        t16a[tid] = acc;
    }
    __syncthreads();
    if (tid < Ec) {
        float acc = in_bv[tid];
        #pragma unroll
        for (int e = 0; e < Ec; ++e) acc += t16a[e] * in_wv[tid * Ec + e];
        t16b[tid] = acc;
    }
    __syncthreads();
    if (tid < Ec) {
        float acc = out_b[tid];
        #pragma unroll
        for (int e = 0; e < Ec; ++e) acc += t16b[e] * out_w[tid * Ec + e];
        t16c[tid] = acc;
    }
    __syncthreads();
    if (tid < NSCc) {
        float acc = o_b[tid];
        #pragma unroll
        for (int e = 0; e < Ec; ++e) acc += t16c[e] * o_w[tid * Ec + e];
        CAVEC[b * NSCc + tid] = acc;
    }
}

// ---------------------------------------------------------------------------
// WMMA conv/GEMM kernel. X: f16 [B][lpx][CIN] (channel-contiguous, position-
// padded if KT>1). W: f16 [KT][COUT][CIN]. Y: f32 [B][L][COUT].
// grid (L/64, COUT/16, B), block 128 (4 waves, 16 positions each).
// A frag: 16-bit A layout (halves 0..7 = K+{0|8}+0..7, halves 8..15 = +16).
// B frag: lane holds col N=lane%16, same K layout as A (halves 0..7 = K+{0|8}+0..7, 8..15 = +16).
// D tile: lane holds pos N=lane%16, 8 contiguous channels -> vector stores.
// ---------------------------------------------------------------------------
template<int CIN, int COUT, int KT, bool ACCUM>
__global__ __launch_bounds__(128)
void k_conv_wmma(const _Float16* __restrict__ X, int lpx,
                 const _Float16* __restrict__ W,
                 const float* __restrict__ bias,
                 const float* __restrict__ extra,   // optional per-batch COUT add
                 float* __restrict__ Y) {
    const int lane = threadIdx.x & 31;
    const int wid  = threadIdx.x >> 5;
    const int n0 = blockIdx.x * 64 + wid * 16;
    const int m0 = blockIdx.y * 16;
    const int b  = blockIdx.z;
    const _Float16* Xb = X + (size_t)b * lpx * CIN;
    const int cbase = m0 + ((lane >> 4) << 3);

    v8f acc;
    #pragma unroll
    for (int r = 0; r < 8; ++r) {
        float a0 = bias ? bias[cbase + r] : 0.f;
        if (extra) a0 += extra[b * COUT + cbase + r];
        acc[r] = a0;
    }

    const int arow = m0 + (lane & 15);
    const int brow = n0 + (lane & 15);
    const int asub = (lane >> 4) << 3;    // +8 K-offset for upper half-wave (A)
    const int bsub = (lane >> 4) << 3;    // +8 K-offset for upper half-wave (B)

    #pragma unroll
    for (int k = 0; k < KT; ++k) {
        const _Float16* Wk = W + (size_t)k * COUT * CIN + (size_t)arow * CIN + asub;
        const _Float16* Xk = Xb + (size_t)(brow + k) * CIN + bsub;
        if (k + 1 < KT) __builtin_prefetch((const void*)(Xk + CIN), 0, 1);
        #pragma unroll
        for (int kb = 0; kb < CIN; kb += 32) {
            v8h alo = *(const v8h*)(Wk + kb);
            v8h ahi = *(const v8h*)(Wk + kb + 16);
            v16h af;
            #pragma unroll
            for (int i = 0; i < 8; ++i) { af[i] = alo[i]; af[8 + i] = ahi[i]; }
            v8h blo = *(const v8h*)(Xk + kb);
            v8h bhi = *(const v8h*)(Xk + kb + 16);
            v16h bf;
            #pragma unroll
            for (int i = 0; i < 8; ++i) { bf[i] = blo[i]; bf[8 + i] = bhi[i]; }
            acc = __builtin_amdgcn_wmma_f32_16x16x32_f16(
                /*neg_a=*/false, af, /*neg_b=*/false, bf,
                /*c_mod=*/(short)0, acc, /*reuse_a=*/false, /*reuse_b=*/false);
        }
    }

    float* Yp = Y + ((size_t)b * Lc + n0 + (lane & 15)) * COUT + cbase;
    if (ACCUM) {
        #pragma unroll
        for (int r = 0; r < 8; ++r) Yp[r] += acc[r];
    } else {
        v4f lo4, hi4;
        #pragma unroll
        for (int r = 0; r < 4; ++r) { lo4[r] = acc[r]; hi4[r] = acc[4 + r]; }
        *(v4f*)(Yp)     = lo4;
        *(v4f*)(Yp + 4) = hi4;
    }
}

// ---------------------------------------------------------------------------
// GroupNorm(1,C) stats per batch over f32 [b][L][C]; stats: {mean, invstd}*B
// grid B, block 256
// ---------------------------------------------------------------------------
__global__ __launch_bounds__(256)
void k_gn_stats(const float* __restrict__ Y, int perB, float* __restrict__ stats) {
    __shared__ float red[256];
    int b = blockIdx.x;
    const float* p = Y + (size_t)b * perB;
    float sum = 0.f, sq = 0.f;
    for (int i = threadIdx.x; i < perB; i += 256) {
        float v = p[i];
        sum += v; sq += v * v;
    }
    float S = block_reduce_256(sum, red);
    float Q = block_reduce_256(sq, red);
    if (threadIdx.x == 0) {
        float m = S / perB;
        float var = Q / perB - m * m;
        stats[2 * b] = m;
        stats[2 * b + 1] = rsqrtf(var + EPSc);
    }
}

// gn + mish -> f16 (same [b][l][c] layout), for stage-1 features
__global__ void k_gn_apply_f16(const float* __restrict__ S, const float* __restrict__ stats,
                               const float* __restrict__ g, const float* __restrict__ beta,
                               _Float16* __restrict__ O, int C, size_t total) {
    size_t idx = (size_t)blockIdx.x * blockDim.x + threadIdx.x;
    if (idx >= total) return;
    int c = (int)(idx % C);
    int b = (int)(idx / ((size_t)Lc * C));
    float v = (S[idx] - stats[2 * b]) * stats[2 * b + 1] * g[c] + beta[c];
    O[idx] = (_Float16)mishf(v);
}

// ---------------------------------------------------------------------------
// LayerNorm over C=192 per (b,l); optionally writes into position-padded f16.
// grid B*L, block 256 (192 active)
// ---------------------------------------------------------------------------
__global__ __launch_bounds__(256)
void k_ln192(const float* __restrict__ X, const float* __restrict__ g,
             const float* __restrict__ beta, _Float16* __restrict__ O,
             int lpo, int pad) {
    __shared__ float red[256];
    int bl = blockIdx.x, tid = threadIdx.x;
    int b = bl / Lc, l = bl % Lc;
    float v = (tid < NSCc) ? X[(size_t)bl * NSCc + tid] : 0.f;
    float S = block_reduce_256(v, red);
    float m = S / NSCc;
    float d = (tid < NSCc) ? (v - m) : 0.f;
    float Q = block_reduce_256(d * d, red);
    float inv = rsqrtf(Q / NSCc + EPSc);
    if (tid < NSCc)
        O[((size_t)b * lpo + l + pad) * NSCc + tid] =
            (_Float16)((v - m) * inv * g[tid] + beta[tid]);
}

// ---------------------------------------------------------------------------
// Flash self-attention, DH=4, H=4. K/V for one (b,h) staged in LDS.
// grid B*H, block 256; AO written f16 into [b][l][32] (upper 16 pre-zeroed).
// ---------------------------------------------------------------------------
__global__ __launch_bounds__(256)
void k_flash(const float* __restrict__ QH, const float* __restrict__ KH,
             const float* __restrict__ VH, _Float16* __restrict__ AO) {
    __shared__ float kk[Lc][4];
    __shared__ float vv[Lc][4];
    int bh = blockIdx.x, b = bh >> 2, h = bh & 3;
    int tid = threadIdx.x;
    for (int i = tid; i < Lc; i += 256) {
        #pragma unroll
        for (int d = 0; d < 4; ++d) {
            kk[i][d] = KH[((size_t)(b * Lc) + i) * Ec + 4 * h + d];
            vv[i][d] = VH[((size_t)(b * Lc) + i) * Ec + 4 * h + d];
        }
    }
    __syncthreads();
    for (int q = tid; q < Lc; q += 256) {
        float q0 = QH[((size_t)(b * Lc) + q) * Ec + 4 * h + 0] * 0.5f;  // DH^-0.5
        float q1 = QH[((size_t)(b * Lc) + q) * Ec + 4 * h + 1] * 0.5f;
        float q2 = QH[((size_t)(b * Lc) + q) * Ec + 4 * h + 2] * 0.5f;
        float q3 = QH[((size_t)(b * Lc) + q) * Ec + 4 * h + 3] * 0.5f;
        float m = -1e30f, s = 0.f, o0 = 0.f, o1 = 0.f, o2 = 0.f, o3 = 0.f;
        for (int j = 0; j < Lc; ++j) {
            float sc = q0 * kk[j][0] + q1 * kk[j][1] + q2 * kk[j][2] + q3 * kk[j][3];
            float nm = fmaxf(m, sc);
            float corr = __expf(m - nm);
            float e = __expf(sc - nm);
            s = s * corr + e;
            o0 = o0 * corr + e * vv[j][0];
            o1 = o1 * corr + e * vv[j][1];
            o2 = o2 * corr + e * vv[j][2];
            o3 = o3 * corr + e * vv[j][3];
            m = nm;
        }
        float inv = 1.f / s;
        size_t base = ((size_t)(b * Lc) + q) * 32 + 4 * h;
        AO[base + 0] = (_Float16)(o0 * inv);
        AO[base + 1] = (_Float16)(o1 * inv);
        AO[base + 2] = (_Float16)(o2 * inv);
        AO[base + 3] = (_Float16)(o3 * inv);
    }
}

// ---------------------------------------------------------------------------
// Final: out[b][o][l] = mish(GN2(s4T)) + resT   (reads [b][l][c] layouts)
// ---------------------------------------------------------------------------
__global__ void k_final(const float* __restrict__ S4T, const float* __restrict__ REST,
                        const float* __restrict__ stats, const float* __restrict__ g,
                        const float* __restrict__ beta, float* __restrict__ out) {
    size_t idx = (size_t)blockIdx.x * blockDim.x + threadIdx.x;
    size_t total = (size_t)Bc * NOUTc * Lc;
    if (idx >= total) return;
    int l = (int)(idx % Lc);
    int o = (int)((idx / Lc) % NOUTc);
    int b = (int)(idx / ((size_t)NOUTc * Lc));
    size_t tidx = ((size_t)b * Lc + l) * NOUTc + o;
    float v = (S4T[tidx] - stats[2 * b]) * stats[2 * b + 1] * g[o] + beta[o];
    out[idx] = mishf(v) + REST[tidx];
}

// ---------------------------------------------------------------------------
// Host launcher
// ---------------------------------------------------------------------------
extern "C" void kernel_launch(void* const* d_in, const int* in_sizes, int n_in,
                              void* d_out, int out_size, void* d_ws, size_t ws_size,
                              hipStream_t stream) {
    // Input index map (setup_inputs dict order, params flattened recursively)
    const float* x          = (const float*)d_in[0];
    const int*   t          = (const int*)  d_in[1];
    const float* text       = (const float*)d_in[2];
    const float* time_table = (const float*)d_in[3];
    const float* t_fc1_w = (const float*)d_in[4];  const float* t_fc1_b = (const float*)d_in[5];
    const float* t_fc2_w = (const float*)d_in[6];  const float* t_fc2_b = (const float*)d_in[7];
    const float* pre_w  = (const float*)d_in[8];   const float* pre_b  = (const float*)d_in[9];
    const float* sc_w   = (const float*)d_in[10];  const float* sc_b   = (const float*)d_in[11];
    const float* post_w = (const float*)d_in[12];  const float* post_b = (const float*)d_in[13];
    const float* res_w  = (const float*)d_in[14];  const float* res_b  = (const float*)d_in[15];
    const float* n1_g = (const float*)d_in[16];    const float* n1_b = (const float*)d_in[17];
    const float* n2_g = (const float*)d_in[18];    const float* n2_b = (const float*)d_in[19];
    const float* an_g = (const float*)d_in[20];    const float* an_b = (const float*)d_in[21];
    // sa: 22..39
    const float* sa_ln_g = (const float*)d_in[22]; const float* sa_ln_b = (const float*)d_in[23];
    const float* sa_qw = (const float*)d_in[24];   const float* sa_qb = (const float*)d_in[25];
    const float* sa_kw = (const float*)d_in[26];   const float* sa_kb = (const float*)d_in[27];
    const float* sa_vw = (const float*)d_in[28];   const float* sa_vb = (const float*)d_in[29];
    const float* sa_ow = (const float*)d_in[30];   const float* sa_ob = (const float*)d_in[31];
    const float* sa_iwq = (const float*)d_in[32];  const float* sa_ibq = (const float*)d_in[33];
    const float* sa_iwk = (const float*)d_in[34];  const float* sa_ibk = (const float*)d_in[35];
    const float* sa_iwv = (const float*)d_in[36];  const float* sa_ibv = (const float*)d_in[37];
    const float* sa_outw = (const float*)d_in[38]; const float* sa_outb = (const float*)d_in[39];
    // ca: 40..61
    const float* ca_vw = (const float*)d_in[46];   const float* ca_vb = (const float*)d_in[47];
    const float* ca_ow = (const float*)d_in[48];   const float* ca_ob = (const float*)d_in[49];
    const float* ca_iwv = (const float*)d_in[54];  const float* ca_ibv = (const float*)d_in[55];
    const float* ca_outw = (const float*)d_in[56]; const float* ca_outb = (const float*)d_in[57];
    const float* ca_vcw = (const float*)d_in[58];  const float* ca_vcb = (const float*)d_in[59];
    const float* ca_cng = (const float*)d_in[60];  const float* ca_cnb = (const float*)d_in[61];
    (void)in_sizes; (void)n_in; (void)out_size;

    float* out = (float*)d_out;
    char* ws = (char*)d_ws;

    // Workspace layout (256B aligned)
    size_t off = 0;
    auto alloc = [&](size_t bytes) { size_t o = off; off = (off + bytes + 255) & ~(size_t)255; return o; };
    size_t oTE    = alloc((size_t)Bc * NINc * 4);
    size_t oXTT   = alloc((size_t)Bc * LPc * NINc * 2);
    size_t oXT    = alloc((size_t)Bc * Lc  * NINc * 2);
    size_t oWPRE  = alloc((size_t)KTAPS * NSCc * NINc * 2);
    size_t oWSC   = alloc((size_t)NSCc * NSCc * 2);
    size_t oWPOST = alloc((size_t)KTAPS * NOUTc * NSCc * 2);
    size_t oWRES  = alloc((size_t)NOUTc * NINc * 2);
    size_t oWQKV  = alloc((size_t)3 * Ec * NSCc * 2);
    size_t oBQKV  = alloc((size_t)3 * Ec * 4);
    size_t oWO    = alloc((size_t)NSCc * 32 * 2);
    size_t oBO    = alloc((size_t)NSCc * 4);
    size_t oCAV   = alloc((size_t)Bc * NSCc * 4);
    size_t oS1T   = alloc((size_t)Bc * Lc * NSCc * 4);
    size_t oS1NT  = alloc((size_t)Bc * Lc * NSCc * 2);
    size_t oS2T   = alloc((size_t)Bc * Lc * NSCc * 4);
    size_t oXN    = alloc((size_t)Bc * Lc * NSCc * 2);
    size_t oQH    = alloc((size_t)Bc * Lc * Ec * 4);
    size_t oKH    = alloc((size_t)Bc * Lc * Ec * 4);
    size_t oVH    = alloc((size_t)Bc * Lc * Ec * 4);
    size_t oAO    = alloc((size_t)Bc * Lc * 32 * 2);
    size_t oS3TP  = alloc((size_t)Bc * LPc * NSCc * 2);
    size_t oS4T   = alloc((size_t)Bc * Lc * NOUTc * 4);
    size_t oREST  = alloc((size_t)Bc * Lc * NOUTc * 4);
    size_t oSTATS = alloc(64 * 4);
    if (ws_size < off) return;   // deterministic no-op if workspace too small

    float*    TE   = (float*)(ws + oTE);
    _Float16* XTT  = (_Float16*)(ws + oXTT);
    _Float16* XT   = (_Float16*)(ws + oXT);
    _Float16* WPRE = (_Float16*)(ws + oWPRE);
    _Float16* WSC  = (_Float16*)(ws + oWSC);
    _Float16* WPOST= (_Float16*)(ws + oWPOST);
    _Float16* WRES = (_Float16*)(ws + oWRES);
    _Float16* WQKV = (_Float16*)(ws + oWQKV);
    float*    BQKV = (float*)(ws + oBQKV);
    _Float16* WO   = (_Float16*)(ws + oWO);
    float*    BO   = (float*)(ws + oBO);
    float*    CAV  = (float*)(ws + oCAV);
    float*    S1T  = (float*)(ws + oS1T);
    _Float16* S1NT = (_Float16*)(ws + oS1NT);
    float*    S2T  = (float*)(ws + oS2T);
    _Float16* XN   = (_Float16*)(ws + oXN);
    float*    QH   = (float*)(ws + oQH);
    float*    KH   = (float*)(ws + oKH);
    float*    VH   = (float*)(ws + oVH);
    _Float16* AO   = (_Float16*)(ws + oAO);
    _Float16* S3TP = (_Float16*)(ws + oS3TP);
    float*    S4T  = (float*)(ws + oS4T);
    float*    REST = (float*)(ws + oREST);
    float*    ST1  = (float*)(ws + oSTATS);
    float*    ST2  = ST1 + 32;

    auto zero = [&](void* p, size_t bytes) {
        size_t n = bytes / 4;
        int grid = (int)((n + 255) / 256);
        if (grid > 65535) grid = 65535;
        k_fill_zero<<<grid, 256, 0, stream>>>((uint32_t*)p, n);
    };
    // Padded/pad-lane buffers must be zero every launch (pads read by WMMA).
    zero(XTT, (size_t)Bc * LPc * NINc * 2);
    zero(S3TP, (size_t)Bc * LPc * NSCc * 2);
    zero(AO, (size_t)Bc * Lc * 32 * 2);

    // 1. time embedding
    k_time_embed<<<Bc, 128, 0, stream>>>(time_table, t, t_fc1_w, t_fc1_b, t_fc2_w, t_fc2_b, TE);

    // 2. weight prep (f16 layouts)
    k_wconv_f16<<<(NSCc * NINc + 255) / 256, 256, 0, stream>>>(pre_w, WPRE, NSCc, NINc, KTAPS);
    k_wconv_f16<<<(NSCc * NSCc + 255) / 256, 256, 0, stream>>>(sc_w, WSC, NSCc, NSCc, 1);
    k_wconv_f16<<<(NOUTc * NSCc + 255) / 256, 256, 0, stream>>>(post_w, WPOST, NOUTc, NSCc, KTAPS);
    k_wconv_f16<<<(NOUTc * NINc + 255) / 256, 256, 0, stream>>>(res_w, WRES, NOUTc, NINc, 1);
    k_attn_wcomb<<<4, 256, 0, stream>>>(sa_qw, sa_qb, sa_kw, sa_kb, sa_vw, sa_vb,
                                        sa_iwq, sa_ibq, sa_iwk, sa_ibk, sa_iwv, sa_ibv,
                                        sa_ow, sa_ob, sa_outw, sa_outb, WQKV, BQKV, WO, BO);
    k_ca_vec<<<Bc, 256, 0, stream>>>(text, ca_vcw, ca_vcb, ca_cng, ca_cnb,
                                     ca_vw, ca_vb, ca_iwv, ca_ibv, ca_outw, ca_outb,
                                     ca_ow, ca_ob, CAV);

    // 3. xt = x + te; transposed f16 (padded) + raw transposed f16
    k_prep_xt<<<dim3(Lc / 32, NINc / 32, Bc), 256, 0, stream>>>(x, TE, XTT, XT);

    // 4. pre conv (K=5, 128->192) via WMMA
    k_conv_wmma<NINc, NSCc, KTAPS, false><<<dim3(Lc / 64, NSCc / 16, Bc), 128, 0, stream>>>(
        XTT, LPc, WPRE, pre_b, nullptr, S1T);

    // 5. GroupNorm(1) + mish -> f16
    k_gn_stats<<<Bc, 256, 0, stream>>>(S1T, Lc * NSCc, ST1);
    {
        size_t total = (size_t)Bc * Lc * NSCc;
        k_gn_apply_f16<<<(int)((total + 255) / 256), 256, 0, stream>>>(S1T, ST1, n1_g, n1_b, S1NT, NSCc, total);
    }

    // 6. 1x1 shortcut conv (192->192) via WMMA
    k_conv_wmma<NSCc, NSCc, 1, false><<<dim3(Lc / 64, NSCc / 16, Bc), 128, 0, stream>>>(
        S1NT, Lc, WSC, sc_b, nullptr, S2T);

    // 7. self-attn: LN -> composed qkv head projections (WMMA) -> flash -> out-proj (WMMA, accum)
    k_ln192<<<Bc * Lc, 256, 0, stream>>>(S2T, sa_ln_g, sa_ln_b, XN, Lc, 0);
    k_conv_wmma<NSCc, Ec, 1, false><<<dim3(Lc / 64, 1, Bc), 128, 0, stream>>>(
        XN, Lc, WQKV + 0 * Ec * NSCc, BQKV + 0, nullptr, QH);
    k_conv_wmma<NSCc, Ec, 1, false><<<dim3(Lc / 64, 1, Bc), 128, 0, stream>>>(
        XN, Lc, WQKV + 1 * Ec * NSCc, BQKV + Ec, nullptr, KH);
    k_conv_wmma<NSCc, Ec, 1, false><<<dim3(Lc / 64, 1, Bc), 128, 0, stream>>>(
        XN, Lc, WQKV + 2 * Ec * NSCc, BQKV + 2 * Ec, nullptr, VH);
    k_flash<<<Bc * Hc, 256, 0, stream>>>(QH, KH, VH, AO);
    // out-proj + self bias + collapsed cross-attention vector, accumulated into S2T
    k_conv_wmma<32, NSCc, 1, true><<<dim3(Lc / 64, NSCc / 16, Bc), 128, 0, stream>>>(
        AO, Lc, WO, BO, CAV, S2T);

    // 8. final LN -> padded f16 for post conv
    k_ln192<<<Bc * Lc, 256, 0, stream>>>(S2T, an_g, an_b, S3TP, LPc, 2);

    // 9. post conv (K=5, 192->256) and residual 1x1 (128->256) via WMMA
    k_conv_wmma<NSCc, NOUTc, KTAPS, false><<<dim3(Lc / 64, NOUTc / 16, Bc), 128, 0, stream>>>(
        S3TP, LPc, WPOST, post_b, nullptr, S4T);
    k_conv_wmma<NINc, NOUTc, 1, false><<<dim3(Lc / 64, NOUTc / 16, Bc), 128, 0, stream>>>(
        XT, Lc, WRES, res_b, nullptr, REST);

    // 10. GroupNorm(1) + mish + residual -> output (B,256,L)
    k_gn_stats<<<Bc, 256, 0, stream>>>(S4T, Lc * NOUTc, ST2);
    {
        size_t total = (size_t)Bc * NOUTc * Lc;
        k_final<<<(int)((total + 255) / 256), 256, 0, stream>>>(S4T, REST, ST2, n2_g, n2_b, out);
    }
}
